// EventDrivenAttention_82583631167656
// MI455X (gfx1250) — matrix operand order repaired and checked by reading, not compiled, not run on hardware
//
#include <hip/hip_runtime.h>
#include <hip/hip_bf16.h>

#define BDIM 2
#define TDIM 64
#define VDIM 32
#define CDIM 512
#define HDIM 8
#define HD   64
#define TVDIM 2048
#define NROW (BDIM * TVDIM)   /* 4096 */
#define C3   (3 * CDIM)       /* 1536 */
#define NEGV (-1e9f)
#define SCALE 0.125f          /* hd^-0.5 = 1/8 */

typedef __attribute__((ext_vector_type(16))) __bf16 v16bf;
typedef __attribute__((ext_vector_type(8)))  __bf16 v8bf;
typedef __attribute__((ext_vector_type(8)))  float  v8f;
typedef __attribute__((ext_vector_type(4)))  int    v4i;

#if __has_builtin(__builtin_amdgcn_global_load_async_to_lds_b128) && \
    __has_builtin(__builtin_amdgcn_s_wait_asynccnt)
#define EDA_ASYNC 1
typedef __attribute__((address_space(1))) v4i gv4i;   // global int4
typedef __attribute__((address_space(3))) v4i lv4i;   // LDS int4
#endif

__device__ __forceinline__ __bf16 f2bf(float f) {
    unsigned u = __builtin_bit_cast(unsigned, f);
    unsigned r = u + 0x7FFFu + ((u >> 16) & 1u);   // round-to-nearest-even
    unsigned short s = (unsigned short)(r >> 16);
    return __builtin_bit_cast(__bf16, s);
}

// A-fragment (16x32, bf16): lane holds row r=lane&15; element j -> K=16*lh+j
__device__ __forceinline__ v16bf load_a_frag(const __bf16* row_base, int lh) {
    v8bf lo = *(const v8bf*)(row_base + 16 * lh);
    v8bf hi = *(const v8bf*)(row_base + 16 * lh + 8);
    v16bf a;
#pragma unroll
    for (int i = 0; i < 8; ++i) { a[i] = lo[i]; a[i + 8] = hi[i]; }
    return a;
}

__device__ __forceinline__ v8f bf16_wmma(v16bf a, v16bf b, v8f c) {
    return __builtin_amdgcn_wmma_f32_16x16x32_bf16(false, a, false, b, (short)0, c, false, false);
}

// ---------------- prep: event weights + query variable mask ----------------
__global__ void eda_prep(const float* __restrict__ di, const int* __restrict__ gmask,
                         const float* __restrict__ w_ev, const float* __restrict__ b_ev,
                         float* __restrict__ ew, float* __restrict__ maskv) {
    int total = BDIM * HDIM * TDIM + BDIM * VDIM;
    for (int idx = blockIdx.x * blockDim.x + threadIdx.x; idx < total;
         idx += blockDim.x * gridDim.x) {
        if (idx < BDIM * HDIM * TDIM) {
            int b = idx / (HDIM * TDIM);
            int r = idx % (HDIM * TDIM);
            int h = r / TDIM;
            int t = r % TDIM;
            float z = b_ev[h];
#pragma unroll
            for (int e = 0; e < 3; ++e)
                z += di[(b * TDIM + t) * 3 + e] * w_ev[e * HDIM + h];
            ew[idx] = 1.0f / (1.0f + __expf(-z));
        } else {
            int j = idx - BDIM * HDIM * TDIM;
            int b = j / VDIM, v = j % VDIM;
            int s = 0;
            for (int k = 0; k < VDIM; ++k) s += gmask[(b * VDIM + v) * VDIM + k];
            maskv[j] = (s > 0) ? 1.0f : 0.0f;
        }
    }
}

// ---------------- dtype conversions / weight transposes ----------------
__global__ void eda_cvt_x(const float* __restrict__ x, __bf16* __restrict__ xh, int n) {
    for (int i = blockIdx.x * blockDim.x + threadIdx.x; i < n; i += blockDim.x * gridDim.x)
        xh[i] = f2bf(x[i]);
}

__global__ void eda_tr_wqkv(const float* __restrict__ w, __bf16* __restrict__ wt) {
    for (int i = blockIdx.x * blockDim.x + threadIdx.x; i < C3 * CDIM;
         i += blockDim.x * gridDim.x) {
        int o = i / CDIM, c = i % CDIM;
        wt[i] = f2bf(w[c * C3 + o]);
    }
}

__global__ void eda_tr_wout(const float* __restrict__ w, __bf16* __restrict__ wt) {
    for (int i = blockIdx.x * blockDim.x + threadIdx.x; i < CDIM * CDIM;
         i += blockDim.x * gridDim.x) {
        int o = i / CDIM, c = i % CDIM;
        wt[i] = f2bf(w[c * CDIM + o]);
    }
}

// ---------------- QKV GEMM: 64x64 macro-tile per wave (4x4 register blocking) ----------------
__global__ void __launch_bounds__(32)
eda_qkv(const __bf16* __restrict__ xh, const __bf16* __restrict__ wqkvT,
        const float* __restrict__ b_qkv,
        __bf16* __restrict__ Q, __bf16* __restrict__ K, __bf16* __restrict__ Vt) {
    int ct0 = blockIdx.x * 64;        // 24 blocks of 64 output cols
    int rt0 = blockIdx.y * 64;        // 64 blocks of 64 rows
    int lane = threadIdx.x;
    int r = lane & 15, lh = lane >> 4;

    v8f acc[4][4] = {};
    for (int kc = 0; kc < CDIM; kc += 32) {
        v16bf a[4], bb[4];
#pragma unroll
        for (int i = 0; i < 4; ++i)
            a[i] = load_a_frag(xh + (size_t)(rt0 + i * 16 + r) * CDIM + kc, lh);
#pragma unroll
        for (int j = 0; j < 4; ++j)
            bb[j] = *(const v16bf*)(wqkvT + (size_t)(ct0 + j * 16 + r) * CDIM + kc + 16 * lh);
#pragma unroll
        for (int i = 0; i < 4; ++i)
#pragma unroll
            for (int j = 0; j < 4; ++j)
                acc[i][j] = bf16_wmma(a[i], bb[j], acc[i][j]);
    }

#pragma unroll
    for (int j = 0; j < 4; ++j) {
        int col = ct0 + j * 16 + r;
        float bias = b_qkv[col];
#pragma unroll
        for (int i = 0; i < 4; ++i) {
#pragma unroll
            for (int e = 0; e < 8; ++e) {
                int row = rt0 + i * 16 + e + 8 * lh;
                int b_ = row / TVDIM, tv = row % TVDIM;
                __bf16 val = f2bf(acc[i][j][e] + bias);
                if (col < CDIM) {
                    int hh = col >> 6, hd = col & 63;
                    Q[(((size_t)b_ * HDIM + hh) * TVDIM + tv) * HD + hd] = val;
                } else if (col < 2 * CDIM) {
                    int c2 = col - CDIM, hh = c2 >> 6, hd = c2 & 63;
                    K[(((size_t)b_ * HDIM + hh) * TVDIM + tv) * HD + hd] = val;
                } else {
                    int c2 = col - 2 * CDIM, hh = c2 >> 6, hd = c2 & 63;
                    Vt[(((size_t)b_ * HDIM + hh) * HD + hd) * TVDIM + tv] = val;
                }
            }
        }
    }
}

// ---------------- flash attention: 4 waves per block, shared K/V tiles in LDS ----------------
__global__ void __launch_bounds__(128)
eda_attn(const __bf16* __restrict__ Q, const __bf16* __restrict__ K,
         const __bf16* __restrict__ Vt, const float* __restrict__ ew,
         const float* __restrict__ maskv, __bf16* __restrict__ Oh) {
    int bid = blockIdx.x;                  // B*H*(TV/64) = 512
    int qb = bid & 31;
    int h  = (bid >> 5) & 7;
    int b  = bid >> 8;
    int tid = threadIdx.x;
    int wid = tid >> 5;
    int lane = tid & 31;
    int r = lane & 15, lh = lane >> 4;
    int q0 = qb * 64 + wid * 16;

    const __bf16* Qp = Q  + ((size_t)(b * HDIM + h) * TVDIM) * HD;
    const __bf16* Kp = K  + ((size_t)(b * HDIM + h) * TVDIM) * HD;
    const __bf16* Vp = Vt + ((size_t)(b * HDIM + h) * HD) * TVDIM;

    __shared__ __align__(16) __bf16 ldsK[32 * 64];      // [key32][hd64]    4 KB
    __shared__ __align__(16) __bf16 ldsV[64 * 32];      // [hd64][key32]    4 KB
    __shared__ __align__(16) __bf16 plds[4][16 * 32];   // per-wave P tile  4 KB

    v16bf aq0 = load_a_frag(Qp + (size_t)(q0 + r) * HD + 0, lh);
    v16bf aq1 = load_a_frag(Qp + (size_t)(q0 + r) * HD + 32, lh);

    float gm[8], gb[8], m_i[8], l_i[8];
#pragma unroll
    for (int i = 0; i < 8; ++i) {
        int row = q0 + i + 8 * lh;
        int t = row / VDIM, vq = row % VDIM;
        float gate = ew[(b * HDIM + h) * TDIM + t];
        float act  = maskv[b * VDIM + vq];
        gm[i] = (act > 0.5f) ? (SCALE * gate) : 0.0f;
        gb[i] = (act > 0.5f) ? ((1.0f - gate) * NEGV) : NEGV;
        m_i[i] = -INFINITY;
        l_i[i] = 0.0f;
    }
    v8f accO[4] = {};

    for (int kt = 0; kt < TVDIM; kt += 32) {
        __syncthreads();   // prior-tile LDS reads done in all waves
        // ---- cooperative stage of K tile (contiguous 4KB) and V tile (64 strided rows) ----
#ifdef EDA_ASYNC
        {
            gv4i* gk = (gv4i*)(void*)(Kp + (size_t)kt * HD);
            lv4i* lk = (lv4i*)(void*)ldsK;
            lv4i* lv = (lv4i*)(void*)ldsV;
#pragma unroll
            for (int s = 0; s < 2; ++s) {
                int idx = tid + s * 128;                 // 256 x 16B = 4KB
                __builtin_amdgcn_global_load_async_to_lds_b128(gk + idx, lk + idx, 0, 0);
                int row = idx >> 2, c = idx & 3;
                gv4i* gvp = (gv4i*)(void*)(Vp + (size_t)row * TVDIM + kt + c * 8);
                __builtin_amdgcn_global_load_async_to_lds_b128(gvp, lv + idx, 0, 0);
            }
            __builtin_amdgcn_s_wait_asynccnt(0);
        }
#else
        {
            const uint4* gk = (const uint4*)(Kp + (size_t)kt * HD);
            uint4* lk = (uint4*)ldsK;
            uint4* lv = (uint4*)ldsV;
#pragma unroll
            for (int s = 0; s < 2; ++s) {
                int idx = tid + s * 128;
                lk[idx] = gk[idx];
                int row = idx >> 2, c = idx & 3;
                lv[idx] = *(const uint4*)(Vp + (size_t)row * TVDIM + kt + c * 8);
            }
        }
#endif
        __syncthreads();   // tiles visible to all waves

        // ---- scores: two 16x16 tiles, contraction over hd=64, K from LDS ----
        v16bf bkA0 = *(const v16bf*)&ldsK[(r) * 64 + 0 + 16 * lh];
        v16bf bkA1 = *(const v16bf*)&ldsK[(r) * 64 + 32 + 16 * lh];
        v16bf bkB0 = *(const v16bf*)&ldsK[(16 + r) * 64 + 0 + 16 * lh];
        v16bf bkB1 = *(const v16bf*)&ldsK[(16 + r) * 64 + 32 + 16 * lh];
        v8f z = {};
        v8f sA = bf16_wmma(aq0, bkA0, bf16_wmma(aq1, bkA1, z));
        v8f sB = bf16_wmma(aq0, bkB0, bf16_wmma(aq1, bkB1, z));

        float alpha[8];
#pragma unroll
        for (int i = 0; i < 8; ++i) {
            float sa = sA[i] * gm[i] + gb[i];
            float sb = sB[i] * gm[i] + gb[i];
            float v = fmaxf(sa, sb);
            v = fmaxf(v, __shfl_xor(v, 8, 32));
            v = fmaxf(v, __shfl_xor(v, 4, 32));
            v = fmaxf(v, __shfl_xor(v, 2, 32));
            v = fmaxf(v, __shfl_xor(v, 1, 32));
            float m_new = fmaxf(m_i[i], v);
            float al = __expf(m_i[i] - m_new);
            float pa = __expf(sa - m_new);
            float pb = __expf(sb - m_new);
            float ps = pa + pb;
            ps += __shfl_xor(ps, 8, 32);
            ps += __shfl_xor(ps, 4, 32);
            ps += __shfl_xor(ps, 2, 32);
            ps += __shfl_xor(ps, 1, 32);
            l_i[i] = l_i[i] * al + ps;
            m_i[i] = m_new;
            alpha[i] = al;
            plds[wid][(i + 8 * lh) * 32 + r]      = f2bf(pa);
            plds[wid][(i + 8 * lh) * 32 + 16 + r] = f2bf(pb);
        }
        // per-wave buffer: HW keeps same-wave DS ops in order; stop compiler reordering
        asm volatile("" ::: "memory");
        v8bf plo = *(const v8bf*)&plds[wid][r * 32 + 16 * lh];
        v8bf phi = *(const v8bf*)&plds[wid][r * 32 + 16 * lh + 8];
        v16bf pf;
#pragma unroll
        for (int i = 0; i < 8; ++i) { pf[i] = plo[i]; pf[i + 8] = phi[i]; }

#pragma unroll
        for (int nt = 0; nt < 4; ++nt) {
            v8f c = accO[nt];
#pragma unroll
            for (int i = 0; i < 8; ++i) c[i] = c[i] * alpha[i];
            v16bf bv = *(const v16bf*)&ldsV[(16 * nt + r) * 32 + 16 * lh];
            accO[nt] = bf16_wmma(pf, bv, c);
        }
    }

#pragma unroll
    for (int nt = 0; nt < 4; ++nt) {
#pragma unroll
        for (int i = 0; i < 8; ++i) {
            float o = accO[nt][i] / l_i[i];
            size_t row = (size_t)b * TVDIM + q0 + i + 8 * lh;
            int c = h * HD + 16 * nt + r;
            Oh[row * CDIM + c] = f2bf(o);
        }
    }
}

// ---------------- output projection: 64x64 macro-tile per wave ----------------
__global__ void __launch_bounds__(32)
eda_outproj(const __bf16* __restrict__ oh, const __bf16* __restrict__ woutT,
            const float* __restrict__ b_out, float* __restrict__ out) {
    int ct0 = blockIdx.x * 64;        // 8 blocks of 64 cols
    int rt0 = blockIdx.y * 64;        // 64 blocks of 64 rows
    int lane = threadIdx.x;
    int r = lane & 15, lh = lane >> 4;

    v8f acc[4][4] = {};
    for (int kc = 0; kc < CDIM; kc += 32) {
        v16bf a[4], bb[4];
#pragma unroll
        for (int i = 0; i < 4; ++i)
            a[i] = load_a_frag(oh + (size_t)(rt0 + i * 16 + r) * CDIM + kc, lh);
#pragma unroll
        for (int j = 0; j < 4; ++j)
            bb[j] = *(const v16bf*)(woutT + (size_t)(ct0 + j * 16 + r) * CDIM + kc + 16 * lh);
#pragma unroll
        for (int i = 0; i < 4; ++i)
#pragma unroll
            for (int j = 0; j < 4; ++j)
                acc[i][j] = bf16_wmma(a[i], bb[j], acc[i][j]);
    }
#pragma unroll
    for (int j = 0; j < 4; ++j) {
        int col = ct0 + j * 16 + r;
        float bias = b_out[col];
#pragma unroll
        for (int i = 0; i < 4; ++i)
#pragma unroll
            for (int e = 0; e < 8; ++e) {
                size_t row = rt0 + i * 16 + e + 8 * lh;
                out[row * CDIM + col] = acc[i][j][e] + bias;
            }
    }
}

extern "C" void kernel_launch(void* const* d_in, const int* in_sizes, int n_in,
                              void* d_out, int out_size, void* d_ws, size_t ws_size,
                              hipStream_t stream) {
    const float* x     = (const float*)d_in[0];
    const float* di    = (const float*)d_in[1];
    const int*   gmask = (const int*)  d_in[2];
    const float* w_qkv = (const float*)d_in[3];
    const float* b_qkv = (const float*)d_in[4];
    const float* w_ev  = (const float*)d_in[5];
    const float* b_ev  = (const float*)d_in[6];
    const float* w_out = (const float*)d_in[7];
    const float* b_out = (const float*)d_in[8];
    float* out = (float*)d_out;

    char* ws = (char*)d_ws;
    size_t off = 0;
    __bf16* Xh     = (__bf16*)(ws + off); off += (size_t)NROW * CDIM * 2;   // 4 MB
    __bf16* WqkvT  = (__bf16*)(ws + off); off += (size_t)C3 * CDIM * 2;     // 1.5 MB
    __bf16* WoutT  = (__bf16*)(ws + off); off += (size_t)CDIM * CDIM * 2;   // 0.5 MB
    __bf16* Qb     = (__bf16*)(ws + off); off += (size_t)NROW * CDIM * 2;   // 4 MB
    __bf16* Kb     = (__bf16*)(ws + off); off += (size_t)NROW * CDIM * 2;   // 4 MB
    __bf16* Vtb    = (__bf16*)(ws + off); off += (size_t)NROW * CDIM * 2;   // 4 MB
    __bf16* Oh     = (__bf16*)(ws + off); off += (size_t)NROW * CDIM * 2;   // 4 MB
    float*  ewbuf  = (float*)(ws + off);  off += (size_t)BDIM * HDIM * TDIM * 4;
    float*  maskv  = (float*)(ws + off);  off += (size_t)BDIM * VDIM * 4;
    (void)ws_size; (void)in_sizes; (void)n_in; (void)out_size;

    eda_prep<<<5, 256, 0, stream>>>(di, gmask, w_ev, b_ev, ewbuf, maskv);
    eda_cvt_x<<<2048, 256, 0, stream>>>(x, Xh, NROW * CDIM);
    eda_tr_wqkv<<<1024, 256, 0, stream>>>(w_qkv, WqkvT);
    eda_tr_wout<<<512, 256, 0, stream>>>(w_out, WoutT);

    eda_qkv<<<dim3(C3 / 64, NROW / 64), 32, 0, stream>>>(Xh, WqkvT, b_qkv, Qb, Kb, Vtb);

    eda_attn<<<BDIM * HDIM * (TVDIM / 64), 128, 0, stream>>>(Qb, Kb, Vtb, ewbuf, maskv, Oh);

    eda_outproj<<<dim3(CDIM / 64, NROW / 64), 32, 0, stream>>>(Oh, WoutT, b_out, out);
}
